// MVG_binaryNetRelaxed_29283087024618
// MI455X (gfx1250) — compile-verified, matrix operand order checked by reading, not run on hardware
//
#include <hip/hip_runtime.h>
#include <hip/hip_bf16.h>

typedef __attribute__((ext_vector_type(16))) _Float16 v16h;
typedef __attribute__((ext_vector_type(8)))  float    v8f;

#define MBATCH 128
#define H      512
#define DIN    784
#define KP0    800           // 784 padded to multiple of 32
#define DOUT   10
#define BETA   0.2f
#define B2     0.04f         // BETA^2
#define PB2    0.06283185307176f   // (pi/2)*BETA^2

// d_out float offsets: h[1280], logprobs[1280], xcov2[128*512*512], loss, frac
#define H_OFF    0
#define LP_OFF   1280
#define XC2_OFF  2560
#define LOSS_OFF (2560 + (size_t)MBATCH * H * H)
#define FRAC_OFF (LOSS_OFF + 1)

// ---------------- WMMA wave-tile helpers (wave32, 16x16 tile) ----------------

__device__ __forceinline__ v8f wmma_f16(v16h a, v16h b, v8f c) {
  // D = A(16x32 f16) * B(32x16 f16) + C(16x16 f32)
  return __builtin_amdgcn_wmma_f32_16x16x32_f16(false, a, false, b, (short)0, c,
                                                false, false);
}

// Fragment loader, contiguous: rows row0..row0+15 of a row-major f16 matrix,
// K-slice k0..k0+31.  Lane l: m = l&15, K-half kb=(l>>4)*8:
//   elems 0..7 = [row0+m][k0+kb .. +7], elems 8..15 = [row0+m][k0+16+kb .. +7]
// Used for A fragments directly, and for B fragments via a transposed copy
// (Bt[n][k] = B[k][n]): both sides become two 16B contiguous loads.
__device__ __forceinline__ v16h load_frag(const _Float16* A, int lda, int row0, int k0,
                                          int lane) {
  v16h a;
  int m = lane & 15, kb = (lane >> 4) << 3;
  const _Float16* p = A + (size_t)(row0 + m) * lda + (k0 + kb);
#pragma unroll
  for (int i = 0; i < 8; ++i) { a[i] = p[i]; a[8 + i] = p[16 + i]; }
  return a;
}

// Row-independent K-slice of a scale vector, fragment-ordered (for packed f16 mul)
__device__ __forceinline__ v16h load_scale_frag(const _Float16* s, int k0, int lane) {
  v16h a;
  int kb = (lane >> 4) << 3;
  const _Float16* p = s + (k0 + kb);
#pragma unroll
  for (int i = 0; i < 8; ++i) { a[i] = p[i]; a[8 + i] = p[16 + i]; }
  return a;
}

// A fragment from f32 source (converted to f16)
__device__ __forceinline__ v16h load_frag_f32(const float* A, int lda, int row0, int k0,
                                              int lane) {
  v16h a;
  int m = lane & 15, kb = (lane >> 4) << 3;
  const float* p = A + (size_t)(row0 + m) * lda + (k0 + kb);
#pragma unroll
  for (int i = 0; i < 8; ++i) {
    a[i]     = (_Float16)p[i];
    a[8 + i] = (_Float16)p[16 + i];
  }
  return a;
}

// ---------------- prep kernels ----------------

__global__ void k_conv_x(const float* __restrict__ x, _Float16* __restrict__ xh) {
  int idx = blockIdx.x * blockDim.x + threadIdx.x;
  if (idx >= MBATCH * KP0) return;
  int r = idx / KP0, j = idx % KP0;
  xh[idx] = (j < DIN) ? (_Float16)x[(size_t)r * DIN + j] : (_Float16)0.0f;
}

// m0t[k][j] = tanh(w0[j][k]/2), K-padded to 800 with zeros (transposed for B frags)
__global__ void k_conv_w0(const float* __restrict__ w0, _Float16* __restrict__ m0t) {
  int idx = blockIdx.x * blockDim.x + threadIdx.x;
  if (idx >= H * KP0) return;
  int k = idx / KP0, j = idx % KP0;
  m0t[idx] = (j < DIN) ? (_Float16)tanhf(0.5f * w0[(size_t)j * H + k]) : (_Float16)0.0f;
}

// m1th[i][j] = m1[j][i]; m1sqt[i][j] = m1[j][i]^2
__global__ void k_conv_w1(const float* __restrict__ w1, _Float16* __restrict__ m1th,
                          _Float16* __restrict__ m1sqt) {
  int idx = blockIdx.x * blockDim.x + threadIdx.x;
  if (idx >= H * H) return;
  int i = idx / H, j = idx % H;
  float m = tanhf(0.5f * w1[(size_t)j * H + i]);
  m1th[idx]  = (_Float16)m;
  m1sqt[idx] = (_Float16)(m * m);
}

// m2t[i][j] = m2[j][i] (f16, B frags); m2f row-major f32 (epilogues/matvecs)
__global__ void k_conv_w2(const float* __restrict__ w2, _Float16* __restrict__ m2t,
                          float* __restrict__ m2f) {
  int idx = blockIdx.x * blockDim.x + threadIdx.x;
  if (idx >= H * H) return;
  int i = idx / H, j = idx % H;
  m2t[idx] = (_Float16)tanhf(0.5f * w2[(size_t)j * H + i]);
  m2f[idx] = tanhf(0.5f * w2[idx]);
}

__global__ void k_conv_wl(const float* __restrict__ wl, float* __restrict__ mlastf) {
  int idx = blockIdx.x * blockDim.x + threadIdx.x;
  if (idx >= H * DOUT) return;
  mlastf[idx] = tanhf(0.5f * wl[idx]);
}

// out[col] = sum_j (1 - tanh(w[j,col]/2)^2), deterministic per-column reduction
__global__ void k_colsum(const float* __restrict__ w, int rows, int cols,
                         float* __restrict__ out) {
  __shared__ float red[256];
  int col = blockIdx.x, tid = threadIdx.x;
  float s = 0.0f;
  for (int j = tid; j < rows; j += 256) {
    float m = tanhf(0.5f * w[(size_t)j * cols + col]);
    s += 1.0f - m * m;
  }
  red[tid] = s;
  __syncthreads();
  for (int st = 128; st > 0; st >>= 1) {
    if (tid < st) red[tid] += red[tid + st];
    __syncthreads();
  }
  if (tid == 0) out[col] = red[0];
}

// ---------------- layer 1: x1bar = tanh(beta*(x@m0+th0)/den), c1 = 1-x1bar^2 ----------------

__global__ void k_layer1(const _Float16* __restrict__ xh, const _Float16* __restrict__ m0t,
                         const float* __restrict__ th0, const float* __restrict__ ds1,
                         _Float16* __restrict__ x1h, float* __restrict__ c1f,
                         _Float16* __restrict__ c1h) {
  int lane = threadIdx.x & 31, wv = threadIdx.x >> 5;
  int id = blockIdx.x * 8 + wv;           // 256 tiles over [128 x 512]
  int mt = id >> 5, nt = id & 31;
  int row0 = mt * 16, col0 = nt * 16;
  v8f acc = {};
  for (int k0 = 0; k0 < KP0; k0 += 32) {
    v16h a = load_frag(xh, KP0, row0, k0, lane);
    v16h b = load_frag(m0t, KP0, col0, k0, lane);   // transposed B
    acc = wmma_f16(a, b, acc);
  }
  int hi = lane >> 4, cl = lane & 15;
  int col = col0 + cl;
  float den = sqrtf(1.0f + PB2 * ds1[col]);
  float t0 = th0[col];
#pragma unroll
  for (int r = 0; r < 8; ++r) {
    int row = row0 + r + 8 * hi;
    float x1 = tanhf(BETA * (acc[r] + t0) / den);
    float c1 = 1.0f - x1 * x1;
    size_t o = (size_t)row * H + col;
    x1h[o] = (_Float16)x1;
    c1f[o] = c1;
    c1h[o] = (_Float16)c1;
  }
}

// ---------------- layer 2 stats: diagsig2 = c1@m1sq + s1, hbar2 = x1@m1 + th1 ----------------

__global__ void k_layer2(const _Float16* __restrict__ c1h, const _Float16* __restrict__ m1sqt,
                         const _Float16* __restrict__ x1h, const _Float16* __restrict__ m1th,
                         const float* __restrict__ s1, const float* __restrict__ th1,
                         _Float16* __restrict__ x2h, float* __restrict__ xc2f,
                         float* __restrict__ d2f) {
  int lane = threadIdx.x & 31, wv = threadIdx.x >> 5;
  int id = blockIdx.x * 8 + wv;
  int mt = id >> 5, nt = id & 31;
  int row0 = mt * 16, col0 = nt * 16;
  v8f accd = {}, acch = {};
  for (int k0 = 0; k0 < H; k0 += 32) {
    v16h a1 = load_frag(c1h, H, row0, k0, lane);
    v16h b1 = load_frag(m1sqt, H, col0, k0, lane);
    accd = wmma_f16(a1, b1, accd);
    v16h a2 = load_frag(x1h, H, row0, k0, lane);
    v16h b2 = load_frag(m1th, H, col0, k0, lane);
    acch = wmma_f16(a2, b2, acch);
  }
  int hi = lane >> 4, cl = lane & 15;
  int col = col0 + cl;
  float s1c = s1[col], t1 = th1[col];
#pragma unroll
  for (int r = 0; r < 8; ++r) {
    int row = row0 + r + 8 * hi;
    float den = sqrtf(1.0f + PB2 * (accd[r] + s1c));
    float x2 = tanhf(BETA * (acch[r] + t1) / den);
    float d2 = 1.0f - x2 * x2;
    size_t o = (size_t)row * H + col;
    x2h[o] = (_Float16)x2;
    xc2f[o] = d2 / den;
    d2f[o] = d2;
  }
}

// ---- xcov2[b] = B2*(m1^T D(c1[b]) m1 + D(s1)) * (xc2⊗xc2) + D(d2) ----
// 2 M-tiles x 4 N-tiles per wave (32x64); A scaled via packed f16 multiply.

__global__ void k_xcov2(const _Float16* __restrict__ m1th, const _Float16* __restrict__ c1h,
                        const float* __restrict__ xc2f, const float* __restrict__ d2f,
                        const float* __restrict__ s1, float* __restrict__ xcov2) {
  int lane = threadIdx.x & 31, wv = threadIdx.x >> 5;
  int id = blockIdx.x * 8 + wv;           // 16384 wave-tiles: b x (16 x 8)
  int b = id >> 7;
  int rem = id & 127;
  int it = rem >> 3, kt = rem & 7;
  int i0 = it * 32, k0c = kt * 64;
  const _Float16* c1row = c1h + (size_t)b * H;
  v8f acc[2][4] = {};
  for (int kk = 0; kk < H; kk += 32) {
    v16h sf = load_scale_frag(c1row, kk, lane);                 // c1[b, j] (f16)
    v16h a0 = load_frag(m1th, H, i0,      kk, lane) * sf;       // m1[j,i]*c1[b,j]
    v16h a1 = load_frag(m1th, H, i0 + 16, kk, lane) * sf;
#pragma unroll
    for (int ni = 0; ni < 4; ++ni) {
      v16h bb = load_frag(m1th, H, k0c + ni * 16, kk, lane);    // m1[j,k]
      acc[0][ni] = wmma_f16(a0, bb, acc[0][ni]);
      acc[1][ni] = wmma_f16(a1, bb, acc[1][ni]);
    }
  }
  int hi = lane >> 4, cl = lane & 15;
  float* outb = xcov2 + (size_t)b * H * H;
#pragma unroll
  for (int ni = 0; ni < 4; ++ni) {
    int col = k0c + ni * 16 + cl;
    float xck = xc2f[(size_t)b * H + col];
    float s1c = s1[col];
    float d2c = d2f[(size_t)b * H + col];
#pragma unroll
    for (int mi = 0; mi < 2; ++mi) {
#pragma unroll
      for (int r = 0; r < 8; ++r) {
        int row = i0 + mi * 16 + r + 8 * hi;
        float sig = acc[mi][ni][r] + (row == col ? s1c : 0.0f);
        float val = B2 * sig * xc2f[(size_t)b * H + row] * xck
                  + (row == col ? d2c : 0.0f);
        outb[(size_t)row * H + col] = val;
      }
    }
  }
}

// ---- diag3 partials: diag(m2^T (xcov2[b] @ m2)) fused into 2x4-blocked GEMM ----

__global__ void k_diag3_gemm(const float* __restrict__ xcov2, const _Float16* __restrict__ m2t,
                             const float* __restrict__ m2f, float* __restrict__ part) {
  int lane = threadIdx.x & 31, wv = threadIdx.x >> 5;
  int id = blockIdx.x * 8 + wv;           // 16384 wave-tiles: b x (16 x 8)
  int b = id >> 7;
  int rem = id & 127;
  int it = rem >> 3, kt = rem & 7;
  int i0 = it * 32, k0c = kt * 64;
  const float* Ab = xcov2 + (size_t)b * H * H;
  v8f acc[2][4] = {};
  for (int kk = 0; kk < H; kk += 32) {
    v16h a0 = load_frag_f32(Ab, H, i0,      kk, lane);
    v16h a1 = load_frag_f32(Ab, H, i0 + 16, kk, lane);
    if (kk + 32 < H) {  // prefetch the streaming xcov2 operand (gfx1250 global_prefetch)
      __builtin_prefetch(Ab + (size_t)(i0 + (lane & 15)) * H + kk + 32, 0, 1);
      __builtin_prefetch(Ab + (size_t)(i0 + 16 + (lane & 15)) * H + kk + 32, 0, 1);
    }
#pragma unroll
    for (int ni = 0; ni < 4; ++ni) {
      v16h bb = load_frag(m2t, H, k0c + ni * 16, kk, lane);
      acc[0][ni] = wmma_f16(a0, bb, acc[0][ni]);
      acc[1][ni] = wmma_f16(a1, bb, acc[1][ni]);
    }
  }
  int hi = lane >> 4, n = lane & 15;
#pragma unroll
  for (int ni = 0; ni < 4; ++ni) {
    int col = k0c + ni * 16 + n;
    float p = 0.0f;
#pragma unroll
    for (int mi = 0; mi < 2; ++mi) {
#pragma unroll
      for (int r = 0; r < 8; ++r) {
        int row = i0 + mi * 16 + r + 8 * hi;
        p += m2f[(size_t)row * H + col] * acc[mi][ni][r];
      }
    }
    p += __shfl_xor(p, 16, 32);           // combine row-halves (deterministic)
    if (lane < 16) part[((size_t)b * 16 + it) * H + col] = p;
  }
}

__global__ void k_diag3_reduce(const float* __restrict__ part, const float* __restrict__ s2,
                               float* __restrict__ diag3) {
  int idx = blockIdx.x * blockDim.x + threadIdx.x;
  if (idx >= MBATCH * H) return;
  int b = idx / H, k = idx % H;
  float s = s2[k];
  const float* p = part + (size_t)b * 16 * H + k;
  for (int it = 0; it < 16; ++it) s += p[(size_t)it * H];
  diag3[idx] = s;
}

// ---------------- layer 3 ----------------

__global__ void k_layer3(const _Float16* __restrict__ x2h, const _Float16* __restrict__ m2t,
                         const float* __restrict__ th2, const float* __restrict__ diag3,
                         float* __restrict__ x3f, float* __restrict__ xc3f,
                         float* __restrict__ d3f) {
  int lane = threadIdx.x & 31, wv = threadIdx.x >> 5;
  int id = blockIdx.x * 8 + wv;
  int mt = id >> 5, nt = id & 31;
  int row0 = mt * 16, col0 = nt * 16;
  v8f acc = {};
  for (int k0 = 0; k0 < H; k0 += 32) {
    v16h a = load_frag(x2h, H, row0, k0, lane);
    v16h b = load_frag(m2t, H, col0, k0, lane);
    acc = wmma_f16(a, b, acc);
  }
  int hi = lane >> 4, cl = lane & 15;
  int col = col0 + cl;
  float t2 = th2[col];
#pragma unroll
  for (int r = 0; r < 8; ++r) {
    int row = row0 + r + 8 * hi;
    size_t o = (size_t)row * H + col;
    float den = sqrtf(1.0f + PB2 * diag3[o]);
    float x3 = tanhf(BETA * (acc[r] + t2) / den);
    float d3 = 1.0f - x3 * x3;
    x3f[o] = x3;
    xc3f[o] = d3 / den;
    d3f[o] = d3;
  }
}

// ---------------- last layer: one block per (b,k); folded quadratic forms ----------------

__device__ __forceinline__ float block_sum256(float v, float* red, int tid) {
  red[tid] = v;
  __syncthreads();
  for (int s = 128; s > 0; s >>= 1) {
    if (tid < s) red[tid] += red[tid + s];
    __syncthreads();
  }
  float r = red[0];
  __syncthreads();
  return r;
}

__global__ void k_lastlayer(const float* __restrict__ xc3f, const float* __restrict__ mlastf,
                            const float* __restrict__ m2f, const float* __restrict__ xcov2,
                            const float* __restrict__ s2, const float* __restrict__ d3f,
                            const float* __restrict__ x3f, const float* __restrict__ slast,
                            const float* __restrict__ thlast, const float* __restrict__ samples,
                            float* __restrict__ hout) {
  __shared__ float w3[H];
  __shared__ float rv[H];
  __shared__ float red[256];
  int b = blockIdx.x / DOUT, k = blockIdx.x % DOUT;
  int tid = threadIdx.x;

  for (int i = tid; i < H; i += 256)
    w3[i] = xc3f[(size_t)b * H + i] * mlastf[(size_t)i * DOUT + k];
  __syncthreads();

  // r = m2 @ w3
  for (int p = tid; p < H; p += 256) {
    float s = 0.0f;
    const float* mr = m2f + (size_t)p * H;
    for (int i = 0; i < H; ++i) s += mr[i] * w3[i];
    rv[p] = s;
  }
  __syncthreads();

  // quad = r^T xcov2[b] r
  float quad = 0.0f;
  const float* xb = xcov2 + (size_t)b * H * H;
  for (int p = tid; p < H; p += 256) {
    float t = 0.0f;
    const float* xr = xb + (size_t)p * H;
    for (int q = 0; q < H; ++q) t += xr[q] * rv[q];
    quad += rv[p] * t;
  }

  float sterm = 0.0f, vterm = 0.0f, hterm = 0.0f;
  for (int i = tid; i < H; i += 256) {
    float w = w3[i];
    sterm += s2[i] * w * w;
    float v = mlastf[(size_t)i * DOUT + k];
    vterm += v * v * d3f[(size_t)b * H + i];
    hterm += x3f[(size_t)b * H + i] * v;
  }

  quad = block_sum256(quad, red, tid);
  sterm = block_sum256(sterm, red, tid);
  vterm = block_sum256(vterm, red, tid);
  hterm = block_sum256(hterm, red, tid);

  if (tid == 0) {
    float dsl = B2 * (quad + sterm) + vterm + slast[k];   // diagsiglast
    float hl = hterm + thlast[k];                          // hlastbar
    dsl = fmaxf(dsl, 0.0f);
    float h = hl * 0.044194173824159216f                   // 1/sqrt(512)
            + sqrtf(dsl * (1.0f / 512.0f)) * samples[(size_t)b * DOUT + k];
    hout[(size_t)b * DOUT + k] = h;
  }
}

// ---------------- log-softmax / nll / argmax / loss ----------------

__global__ void k_loss(const float* __restrict__ hout, const int* __restrict__ target,
                       float* __restrict__ lp, float* __restrict__ lossp,
                       float* __restrict__ fracp) {
  __shared__ float redf[128];
  __shared__ int redi[128];
  int b = threadIdx.x;   // 128 threads, one per batch row
  float hv[DOUT];
  float mx = -1e30f;
  for (int k = 0; k < DOUT; ++k) {
    hv[k] = hout[(size_t)b * DOUT + k];
    mx = fmaxf(mx, hv[k]);
  }
  float se = 0.0f;
  for (int k = 0; k < DOUT; ++k) se += expf(hv[k] - mx);
  float lse = mx + logf(se);
  int best = 0;
  float bv = hv[0];
  for (int k = 0; k < DOUT; ++k) {
    lp[(size_t)b * DOUT + k] = hv[k] - lse;
    if (hv[k] > bv) { bv = hv[k]; best = k; }
  }
  int t = target[b];
  redf[b] = -(hv[t] - lse);
  redi[b] = (best == t) ? 1 : 0;
  __syncthreads();
  for (int s = 64; s > 0; s >>= 1) {
    if (b < s) { redf[b] += redf[b + s]; redi[b] += redi[b + s]; }
    __syncthreads();
  }
  if (b == 0) {
    *lossp = redf[0] / 128.0f;           // nll + 0.0*R (R finite -> contributes 0)
    *fracp = (float)redi[0] / 128.0f;
  }
}

// ---------------- host launch ----------------

extern "C" void kernel_launch(void* const* d_in, const int* in_sizes, int n_in,
                              void* d_out, int out_size, void* d_ws, size_t ws_size,
                              hipStream_t stream) {
  (void)in_sizes; (void)n_in; (void)out_size; (void)ws_size;
  const float* x       = (const float*)d_in[0];
  const float* samples = (const float*)d_in[1];
  const float* w0      = (const float*)d_in[2];
  const float* w1      = (const float*)d_in[3];
  const float* w2      = (const float*)d_in[4];
  const float* wlast   = (const float*)d_in[5];
  const float* th0     = (const float*)d_in[6];
  const float* th1     = (const float*)d_in[7];
  const float* th2     = (const float*)d_in[8];
  const float* thlast  = (const float*)d_in[9];
  const int*   target  = (const int*)d_in[10];

  char* ws = (char*)d_ws;
  size_t off = 0;
  auto alloc = [&](size_t bytes) -> void* {
    void* p = ws + off;
    off = (off + bytes + 255) & ~(size_t)255;
    return p;
  };

  _Float16* xh    = (_Float16*)alloc((size_t)MBATCH * KP0 * 2);
  _Float16* m0t   = (_Float16*)alloc((size_t)H * KP0 * 2);
  _Float16* m1th  = (_Float16*)alloc((size_t)H * H * 2);
  _Float16* m1sqt = (_Float16*)alloc((size_t)H * H * 2);
  _Float16* m2t   = (_Float16*)alloc((size_t)H * H * 2);
  float*    m2f   = (float*)alloc((size_t)H * H * 4);
  float*    mlastf= (float*)alloc((size_t)H * DOUT * 4);
  float*    ds1   = (float*)alloc((size_t)H * 4);
  float*    s1    = (float*)alloc((size_t)H * 4);
  float*    s2    = (float*)alloc((size_t)H * 4);
  float*    slast = (float*)alloc((size_t)DOUT * 4);
  _Float16* x1h   = (_Float16*)alloc((size_t)MBATCH * H * 2);
  _Float16* c1h   = (_Float16*)alloc((size_t)MBATCH * H * 2);
  float*    c1f   = (float*)alloc((size_t)MBATCH * H * 4);  // kept for layout stability
  _Float16* x2h   = (_Float16*)alloc((size_t)MBATCH * H * 2);
  float*    xc2f  = (float*)alloc((size_t)MBATCH * H * 4);
  float*    d2f   = (float*)alloc((size_t)MBATCH * H * 4);
  float*    diag3 = (float*)alloc((size_t)MBATCH * H * 4);
  float*    x3f   = (float*)alloc((size_t)MBATCH * H * 4);
  float*    xc3f  = (float*)alloc((size_t)MBATCH * H * 4);
  float*    d3f   = (float*)alloc((size_t)MBATCH * H * 4);
  float*    part  = (float*)alloc((size_t)MBATCH * 16 * H * 4);
  (void)c1f;

  float* out   = (float*)d_out;
  float* hptr  = out + H_OFF;
  float* lpptr = out + LP_OFF;
  float* xcov2 = out + XC2_OFF;
  float* lossp = out + LOSS_OFF;
  float* fracp = out + FRAC_OFF;

  // prep: weight transforms (transposed f16 copies) + deterministic column sums
  k_conv_x <<<(MBATCH * KP0 + 255) / 256, 256, 0, stream>>>(x, xh);
  k_conv_w0<<<(H * KP0 + 255) / 256, 256, 0, stream>>>(w0, m0t);
  k_conv_w1<<<(H * H + 255) / 256, 256, 0, stream>>>(w1, m1th, m1sqt);
  k_conv_w2<<<(H * H + 255) / 256, 256, 0, stream>>>(w2, m2t, m2f);
  k_conv_wl<<<(H * DOUT + 255) / 256, 256, 0, stream>>>(wlast, mlastf);
  k_colsum<<<H, 256, 0, stream>>>(w0, DIN, H, ds1);
  k_colsum<<<H, 256, 0, stream>>>(w1, H, H, s1);
  k_colsum<<<H, 256, 0, stream>>>(w2, H, H, s2);
  k_colsum<<<DOUT, 256, 0, stream>>>(wlast, H, DOUT, slast);

  // network
  k_layer1<<<32, 256, 0, stream>>>(xh, m0t, th0, ds1, x1h, c1f, c1h);
  k_layer2<<<32, 256, 0, stream>>>(c1h, m1sqt, x1h, m1th, s1, th1, x2h, xc2f, d2f);
  k_xcov2<<<2048, 256, 0, stream>>>(m1th, c1h, xc2f, d2f, s1, xcov2);
  k_diag3_gemm<<<2048, 256, 0, stream>>>(xcov2, m2t, m2f, part);
  k_diag3_reduce<<<(MBATCH * H + 255) / 256, 256, 0, stream>>>(part, s2, diag3);
  k_layer3<<<32, 256, 0, stream>>>(x2h, m2t, th2, diag3, x3f, xc3f, d3f);
  k_lastlayer<<<MBATCH * DOUT, 256, 0, stream>>>(xc3f, mlastf, m2f, xcov2, s2, d3f,
                                                 x3f, slast, thlast, samples, hptr);
  k_loss<<<1, 128, 0, stream>>>(hptr, target, lpptr, lossp, fracp);
}
